// ALIF_86096914416146
// MI455X (gfx1250) — matrix operand order repaired and checked by reading, not compiled, not run on hardware
//
#include <hip/hip_runtime.h>

// ---------------------------------------------------------------------------
// ALIF spiking RNN forward, persistent-kernel formulation for MI455X (gfx1250).
//
//  - 500 sequential steps => persistent kernel + 1 grid barrier per step.
//  - rec = spk([64x1024],{0,1}) x W^T([1024x1024]) via v_wmma_f32_16x16x32_bf16.
//  - W converted to bf16 once (row-major row i == B-fragment column i: the
//    contiguous-K (k = 16h + s, same as A) B layout used here reads 32 contiguous bytes per lane, so no
//    transpose needed). Spikes are exactly representable in bf16.
//  - Per-WG W slice (32 rows, 64 KB) staged in LDS once; full spike matrix
//    (128 KB) re-staged each step with GLOBAL_LOAD_ASYNC_TO_LDS_B128
//    (ASYNCcnt). 192 KB LDS of the 320 KB/WGP budget.
//  - Neuron state (mem/refac/v_th/a/spk) lives in VGPRs for all 500 steps.
//  - Spike carry double-buffered in d_ws (hot in 192 MB L2) so one barrier
//    per step suffices (no read/write race between WGs).
// ---------------------------------------------------------------------------

#define TSTEPS   500
#define TLAT     2
#define BATCH    64
#define NN       1024
#define NSLICE   32                 // neurons per workgroup
#define NWG      (NN / NSLICE)      // 32 persistent workgroups
#define THREADS  256                // 8 waves (wave32)

typedef __attribute__((ext_vector_type(16))) __bf16 v16bf;
typedef __attribute__((ext_vector_type(8)))  __bf16 v8bf;
typedef __attribute__((ext_vector_type(8)))  float  v8f;
typedef __attribute__((ext_vector_type(4)))  float  v4f;

// 16-byte payload type matching the async-LDS builtin's parameter type
typedef int b128_t __attribute__((vector_size(4 * sizeof(int))));
typedef __attribute__((address_space(1))) b128_t* b128_gptr;
typedef __attribute__((address_space(3))) b128_t* b128_lptr;

#if defined(__AMDGCN__) && __has_builtin(__builtin_amdgcn_global_load_async_to_lds_b128)
#define HAVE_ASYNC_LDS 1
#else
#define HAVE_ASYNC_LDS 0
#endif

__device__ __forceinline__ void async_copy_b128(const void* g, void* l) {
#if HAVE_ASYNC_LDS
    __builtin_amdgcn_global_load_async_to_lds_b128(
        (b128_gptr)(void*)g,
        (b128_lptr)l,
        0 /*imm offset*/, 0 /*cpol*/);
#else
    *(v4f*)l = *(const v4f*)g;
#endif
}

__device__ __forceinline__ void wait_async_copies() {
#if HAVE_ASYNC_LDS
#if __has_builtin(__builtin_amdgcn_s_wait_asynccnt)
    __builtin_amdgcn_s_wait_asynccnt(0);
#else
    asm volatile("s_wait_asynccnt 0x0" ::: "memory");
#endif
#endif
}

// Device-wide barrier: cumulative counter (500*32 = 16000 << 2^32, no reset
// race). RELEASE on arrive publishes this step's spike stores at agent scope;
// ACQUIRE on the spin makes them visible before the next step's staging.
__device__ __forceinline__ void grid_sync(unsigned* cnt, unsigned target) {
    __syncthreads();
    if (threadIdx.x == 0) {
        __hip_atomic_fetch_add(cnt, 1u, __ATOMIC_RELEASE, __HIP_MEMORY_SCOPE_AGENT);
        while (__hip_atomic_load(cnt, __ATOMIC_ACQUIRE, __HIP_MEMORY_SCOPE_AGENT) < target) {
            __builtin_amdgcn_s_sleep(8);
        }
    }
    __syncthreads();
}

// ---------------------------------------------------------------------------
// Init: W fp32 -> bf16, zero both spike carry buffers, zero barrier counter.
// Re-run on every launch (graph replays re-zero the counter deterministically).
// ---------------------------------------------------------------------------
extern "C" __global__ void alif_init(const float* __restrict__ W,
                                     __bf16* __restrict__ Wbf,
                                     __bf16* __restrict__ s0,
                                     __bf16* __restrict__ s1,
                                     unsigned* __restrict__ cnt) {
    size_t i = (size_t)blockIdx.x * blockDim.x + threadIdx.x;
    if (i < (size_t)NN * NN) Wbf[i] = (__bf16)W[i];
    if (i < (size_t)BATCH * NN) { s0[i] = (__bf16)0.0f; s1[i] = (__bf16)0.0f; }
    if (i == 0) *cnt = 0u;
}

// ---------------------------------------------------------------------------
// Persistent ALIF kernel. Grid = 32 WGs x 256 threads. Dyn LDS = 192 KB:
//   sW   [32][1024] bf16  (this WG's W rows, staged once)
//   sSpk [64][1024] bf16  (all spikes, re-staged each step)
// Wave w: batch tile (w&3)*16, neuron tile (w>>2)*16 within the slice.
// ---------------------------------------------------------------------------
extern "C" __global__ void __launch_bounds__(THREADS, 1)
alif_persistent(const float* __restrict__ x,       // [T][B][N]
                const float* __restrict__ beta_p,  // [N]
                const float* __restrict__ p_p,     // [N]
                const float* __restrict__ b_p,     // [N]
                float*       __restrict__ out,     // [T][B][N]
                const __bf16* __restrict__ Wbf,    // [N][N]
                __bf16* __restrict__ spkA,         // carry buf 0
                __bf16* __restrict__ spkB,         // carry buf 1
                unsigned* __restrict__ cnt) {
    extern __shared__ char smem[];
    __bf16* sW   = (__bf16*)smem;                                  // 64 KB
    __bf16* sSpk = (__bf16*)(smem + (size_t)NSLICE * NN * sizeof(__bf16)); // 128 KB

    const int tid   = threadIdx.x;
    const int wave  = tid >> 5;
    const int lane  = tid & 31;
    const int lo16  = lane & 15;
    const int hi    = lane >> 4;           // 0 or 1 (half-wave)
    const int bTile = (wave & 3) * 16;     // batch rows of this wave's tile
    const int nTile = (wave >> 2) * 16;    // neuron base within WG slice
    const int iGlob = blockIdx.x * NSLICE + nTile + lo16;  // this lane's neuron

    // ---- stage W slice once: 32 rows x 1024 bf16 = 64 KB (16 x 4 KB passes)
    {
        const char* gw = (const char*)(Wbf + (size_t)blockIdx.x * NSLICE * NN);
        char*       lw = (char*)sW;
        #pragma unroll
        for (int it = 0; it < (NSLICE * NN * 2) / (THREADS * 16); ++it) {
            int off = (it * THREADS + tid) * 16;
            async_copy_b128(gw + off, lw + off);
        }
    }

    // ---- per-neuron clipped params (constant across all steps)
    float beta = fminf(fmaxf(beta_p[iGlob], 0.001f), 0.999f);
    float pp   = fminf(fabsf(p_p[iGlob]), 0.999f);
    float bb   = fminf(fmaxf(fabsf(b_p[iGlob]), 0.001f), 1.0f);

    // ---- neuron state in registers: element e -> (batch bTile + hi*8 + e, neuron iGlob)
    float mem[8], refac[8], vth[8], aad[8], spk[8];
    #pragma unroll
    for (int e = 0; e < 8; ++e) {
        mem[e] = 0.0f; refac[e] = (float)TLAT; vth[e] = 1.0f;
        aad[e] = 0.0f; spk[e] = 0.0f;
    }

    wait_async_copies();
    __syncthreads();

    const __bf16* aRow = sSpk + (size_t)(bTile + lo16) * NN;  // A: spike row (M=batch)
    const __bf16* bRow = sW   + (size_t)(nTile + lo16) * NN;  // B: W row (N=neuron)

    for (int t = 0; t < TSTEPS; ++t) {
        // double buffer: read (t-1)'s spikes, write t's
        const __bf16* gRead  = (t & 1) ? spkA : spkB;
        __bf16*       gWrite = (t & 1) ? spkB : spkA;

        // ---- stage all spikes: 64x1024 bf16 = 128 KB (32 x 4 KB async passes)
        {
            const char* gs = (const char*)gRead;
            char*       ls = (char*)sSpk;
            #pragma unroll
            for (int it = 0; it < (BATCH * NN * 2) / (THREADS * 16); ++it) {
                int off = (it * THREADS + tid) * 16;
                async_copy_b128(gs + off, ls + off);
            }
        }
        wait_async_copies();
        __syncthreads();

        // ---- rec = spk x W^T for this wave's 16x16 tile; K = 1024 in 32 WMMAs,
        //      4 interleaved accumulators to break the D->C dependence chain.
        v8f acc[4] = {};
        #pragma unroll
        for (int kk = 0; kk < NN / 32; ++kk) {
            // A fragment (16x32 bf16): lane holds row M=lo16,
            // 16 contiguous K at kk*32 + hi*16
            v16bf af = *(const v16bf*)(aRow + kk * 32 + hi * 16);
            // B fragment (32x16 bf16): lane holds col N=lo16,
            // 16 contiguous K at kk*32 + hi*16  == 32 contiguous bytes of W row
            v16bf bf = *(const v16bf*)(bRow + kk * 32 + hi * 16);
            acc[kk & 3] = __builtin_amdgcn_wmma_f32_16x16x32_bf16(
                false, af, false, bf, (short)0, acc[kk & 3], false, false);
        }

        // ---- elementwise ALIF dynamics + outputs
        const float gate = (t >= TLAT) ? 1.0f : 0.0f;
        const size_t tBase = (size_t)t * BATCH * NN;
        #pragma unroll
        for (int e = 0; e < 8; ++e) {
            int   m   = bTile + hi * 8 + e;                       // batch row
            float rec = acc[0][e] + acc[1][e] + acc[2][e] + acc[3][e];
            float cur = x[tBase + (size_t)m * NN + iGlob] + gate * rec;
            if (spk[e] > 0.0f)        refac[e] = 0.0f;
            if (refac[e] < (float)TLAT) cur = 0.0f;
            refac[e] += 1.0f;
            float nm = mem[e] * beta + cur;
            float s  = (nm - vth[e]) > 0.0f ? 1.0f : 0.0f;
            mem[e] = nm * (1.0f - s);
            aad[e] = pp * aad[e] + s;
            vth[e] = 1.0f + bb * aad[e];
            spk[e] = s;
            out[tBase + (size_t)m * NN + iGlob] = s;              // fp32 result
            gWrite[(size_t)m * NN + iGlob] = (__bf16)s;           // bf16 carry
        }

        grid_sync(cnt, (unsigned)((t + 1) * NWG));
    }
}

// ---------------------------------------------------------------------------
// Launch. d_ws layout: [0, 2MB) W bf16 | [+128KB) spike buf0 | [+128KB) buf1 |
// barrier counter. Everything on `stream`, graph-capture safe.
// ---------------------------------------------------------------------------
extern "C" void kernel_launch(void* const* d_in, const int* in_sizes, int n_in,
                              void* d_out, int out_size, void* d_ws, size_t ws_size,
                              hipStream_t stream) {
    const float* x    = (const float*)d_in[0];
    const float* W    = (const float*)d_in[1];
    const float* betp = (const float*)d_in[2];
    const float* pp   = (const float*)d_in[3];
    const float* bp   = (const float*)d_in[4];
    float* out = (float*)d_out;

    char* ws = (char*)d_ws;
    __bf16*   Wbf = (__bf16*)ws;
    __bf16*   s0  = (__bf16*)(ws + (size_t)NN * NN * 2);
    __bf16*   s1  = (__bf16*)(ws + (size_t)NN * NN * 2 + (size_t)BATCH * NN * 2);
    unsigned* cnt = (unsigned*)(ws + (size_t)NN * NN * 2 + 2 * (size_t)BATCH * NN * 2);

    alif_init<<<(NN * NN) / 256, 256, 0, stream>>>(W, Wbf, s0, s1, cnt);

    size_t smem = (size_t)(NSLICE * NN + BATCH * NN) * sizeof(__bf16); // 192 KB
    (void)hipFuncSetAttribute((const void*)alif_persistent,
                              hipFuncAttributeMaxDynamicSharedMemorySize,
                              (int)smem);
    alif_persistent<<<NWG, THREADS, smem, stream>>>(x, betp, pp, bp, out,
                                                    Wbf, s0, s1, cnt);
}
